// ScientistBrain_15195594293750
// MI455X (gfx1250) — matrix operand order repaired and checked by reading, not compiled, not run on hardware
//
#include <hip/hip_runtime.h>
#include <math.h>

// ---------------- problem constants ----------------
#define Bsz   64
#define Tsz   2048
#define Hsz   256
#define INsz  8
#define Lsz   4
#define NPsz  2

// d_out layout (return-order concat, all f32):
//   predictions [B, T-1, IN] | hidden [B,T,H] | anomaly [B,T] | surprise [B,T-1] | properties [B,NP]
#define PRED_OFF  0
#define HID_OFF   ((size_t)Bsz*(Tsz-1)*INsz)
#define ANOM_OFF  (HID_OFF + (size_t)Bsz*Tsz*Hsz)
#define SURP_OFF  (ANOM_OFF + (size_t)Bsz*Tsz)
#define PROP_OFF  (SURP_OFF + (size_t)Bsz*(Tsz-1))

typedef _Float16 f16;
typedef __attribute__((ext_vector_type(16))) _Float16 v16h;
typedef __attribute__((ext_vector_type(8)))  float    v8f;

// ---------------- CDNA5 data movers ----------------
// Preferred: Tensor Data Mover (TENSORcnt).  Fallback: async global->LDS
// (ASYNCcnt) via inline asm, which is already probe-verified to assemble.
#if defined(__has_builtin)
#if __has_builtin(__builtin_amdgcn_tensor_load_to_lds)
#define HAVE_TDM 1
#endif
#endif
#define ASYNC_COPY 1

#ifdef HAVE_TDM
typedef unsigned int tdm_v4u __attribute__((ext_vector_type(4)));
typedef int          tdm_v8i __attribute__((ext_vector_type(8)));
typedef int          tdm_v4i __attribute__((ext_vector_type(4)));

// One-shot TDM descriptor: copy 64 KB (16384 f32, 1-D tile) from global
// `gsrc` into LDS byte offset `lds_off`.  D# packing per ISA ch.8:
//  g0: [1:0]=count=1 | [63:32]=lds_addr | [120:64]=global_addr | [127:126]=type=2
//  g1: [17:16]=data_size(4B) | [79:48]=tensor_dim0 | [111:80]=tensor_dim1
//      [127:112]=tile_dim0 | [143:128]=tile_dim1(0) | [207:160]=dim0_stride
// This toolchain's builtin is the 6-arg form:
//   (uint32x4 g0, int32x8 g1, int32x4, int32x4, int32x8, i32 cpol)
static __device__ __forceinline__ void tdm_load_64k(const void* gsrc,
                                                    unsigned lds_off) {
  unsigned long long ga = (unsigned long long)(uintptr_t)gsrc;
  tdm_v4u g0 = { 1u,
                 lds_off,
                 (unsigned)(ga & 0xFFFFFFFFu),
                 (unsigned)(((ga >> 32) & 0x01FFFFFFu) | 0x80000000u) };
  tdm_v8i g1 = { (int)(2u << 16),          // data_size = 4 bytes
                 (int)(0x4000u << 16),     // tensor_dim0 = 16384 (lo16)
                 (int)(1u << 16),          // tensor_dim1 = 1
                 (int)(0x4000u << 16),     // tile_dim0 = 16384
                 0,                        // tile_dim1 = tile_dim2 = 0
                 16384,                    // tensor_dim0_stride
                 0, 0 };
  tdm_v4i gz4 = { 0, 0, 0, 0 };
  tdm_v8i gz8 = { 0, 0, 0, 0, 0, 0, 0, 0 };
  __builtin_amdgcn_tensor_load_to_lds(g0, g1, gz4, gz4, gz8, 0);
}
static __device__ __forceinline__ void wait_tensor0() {
#if __has_builtin(__builtin_amdgcn_s_wait_tensorcnt)
  __builtin_amdgcn_s_wait_tensorcnt(0);
#else
  asm volatile("s_wait_tensorcnt 0" ::: "memory");
#endif
}
#endif  // HAVE_TDM

// Per-lane async copy (16 B/lane), tracked by ASYNCcnt.  VDST operand is the
// per-lane LDS byte address (low 32 bits of a generic LDS pointer).
static __device__ __forceinline__ void copy16_to_lds(void* lds, const void* g) {
#if ASYNC_COPY
  asm volatile("global_load_async_to_lds_b128 %0, %1, off"
               :: "v"((unsigned int)(uintptr_t)lds), "v"(g)
               : "memory");
#else
  *(uint4*)lds = *(const uint4*)g;
#endif
}
static __device__ __forceinline__ void wait_async0() {
#if ASYNC_COPY
  asm volatile("s_wait_asynccnt 0" ::: "memory");
#endif
}

// Stage one 64 KB U-step into LDS; `tid` in [0,256).  Issue + completion-wait
// are split so the copy can overlap compute.
static __device__ __forceinline__ void stage_U_issue(float* Ub, const float* g,
                                                     int tid) {
#ifdef HAVE_TDM
  if (tid < 32)  // single TDM descriptor from wave 0 (EXEC ignored by TDM)
    tdm_load_64k(g, (unsigned)(uintptr_t)Ub);
#else
#pragma unroll
  for (int i = 0; i < 16; ++i)
    copy16_to_lds((char*)Ub + tid*16 + i*4096,
                  (const char*)g + tid*16 + i*4096);
#endif
}
static __device__ __forceinline__ void stage_U_wait() {
#ifdef HAVE_TDM
  wait_tensor0();   // no-op for waves with TENSORcnt==0
#else
  wait_async0();
#endif
}

// ---- WMMA fragment loaders (wave32, 16x16x32 f16, per CDNA5 ISA layouts) ----
// A (16x32 f16): lane l -> row M=l&15, kh=l>>4; elems 0..7 = K=8*kh+e,
// elems 8..15 = K=16+8*kh+(e-8)  => two contiguous 16-byte chunks.
static __device__ __forceinline__ v16h loadA_frag(const f16* rowp, int k0) {
  union { v16h v; uint4 q[2]; } u;
  u.q[0] = *(const uint4*)(rowp + k0);
  u.q[1] = *(const uint4*)(rowp + k0 + 16);
  return u.v;
}
// B (32x16 f16) from K-contiguous (transposed) storage: lane l -> col N=l&15,
// kgrp=l>>4; elems 0..7 = K=8*kgrp+e, elems 8..15 = K=16+8*kgrp+(e-8).
static __device__ __forceinline__ v16h loadB_frag(const f16* colp, int k0) {
  union { v16h v; uint4 q[2]; } u;
  u.q[0] = *(const uint4*)(colp + k0);
  u.q[1] = *(const uint4*)(colp + k0 + 16);
  return u.v;
}

// ---------------- input projection: hseq[t*B+b][h] = x[b,t,:] @ in_w + in_b ----
__global__ __launch_bounds__(256) void lnn_inproj(
    const float* __restrict__ x, const float* __restrict__ in_w,
    const float* __restrict__ in_b, float* __restrict__ hseq) {
  int m = blockIdx.x;            // m = t*B + b
  int h = threadIdx.x;
  int t = m >> 6, b = m & 63;
  const float* xp = x + ((size_t)b*Tsz + t)*INsz;
  float acc = in_b[h];
#pragma unroll
  for (int k = 0; k < INsz; ++k) acc += xp[k] * in_w[k*Hsz + h];
  hseq[(size_t)m*Hsz + h] = acc;
}

// ------------- batched GEMM: U[m][n] = sum_k hseq[m][k] * Wx[k][n] -------------
// Persistent blocks: WxT staged once, then 8 chunks of 64 rows each.
#define GCHUNK 8
__global__ __launch_bounds__(256) void lnn_gemm_hWx(
    const float* __restrict__ A, const float* __restrict__ Wx,
    float* __restrict__ U) {
  extern __shared__ char smem[];
  f16* WxT = (f16*)smem;                 // [N=256][K=256] = Wx transposed, 128 KB
  f16* Ah  = (f16*)(smem + 131072);      // [64][256] f16, 32 KB
  const int tid = threadIdx.x;
  for (int i = tid; i < Hsz*Hsz; i += 256) {
    int k = i >> 8, n = i & 255;
    WxT[n*Hsz + k] = (f16)Wx[i];
  }

  const int wv = tid >> 5, lane = tid & 31;
  const int mloc = lane & 15, kh = lane >> 4;
  const int n0 = wv * 32;                // this wave owns n-tiles 2*wv, 2*wv+1

  for (int ch = 0; ch < GCHUNK; ++ch) {
    const int m0 = (blockIdx.x * GCHUNK + ch) * 64;
    if (ch + 1 < GCHUNK)  // warm next chunk's A rows
      __builtin_prefetch(A + (size_t)(m0 + 64)*Hsz + tid*64, 0, 1);
    __syncthreads();       // previous chunk done reading Ah (and WxT staged, ch==0)
    for (int i = tid; i < 64*Hsz; i += 256) {
      int r = i >> 8, c = i & 255;
      Ah[i] = (f16)A[(size_t)(m0 + r)*Hsz + c];
    }
    __syncthreads();

    v8f acc[4][2];
#pragma unroll
    for (int mt = 0; mt < 4; ++mt)
#pragma unroll
      for (int j = 0; j < 2; ++j)
#pragma unroll
        for (int r = 0; r < 8; ++r) acc[mt][j][r] = 0.0f;

#pragma unroll
    for (int kb = 0; kb < 8; ++kb) {
      const int kbase = kb * 32;
      v16h a[4], bf[2];
#pragma unroll
      for (int mt = 0; mt < 4; ++mt)
        a[mt] = loadA_frag(Ah + (mt*16 + mloc)*Hsz, kbase + kh*8);
#pragma unroll
      for (int j = 0; j < 2; ++j)
        bf[j] = loadB_frag(WxT + (n0 + j*16 + mloc)*Hsz, kbase + kh*8);
#pragma unroll
      for (int mt = 0; mt < 4; ++mt)
#pragma unroll
        for (int j = 0; j < 2; ++j)
          acc[mt][j] = __builtin_amdgcn_wmma_f32_16x16x32_f16(
              false, a[mt], false, bf[j], (short)0, acc[mt][j], false, false);
    }

#pragma unroll
    for (int mt = 0; mt < 4; ++mt)
#pragma unroll
      for (int j = 0; j < 2; ++j) {
        int n = n0 + j*16 + (lane & 15);
#pragma unroll
        for (int r = 0; r < 8; ++r) {
          int m = mt*16 + r + 8*kh;
          U[(size_t)(m0 + m)*Hsz + n] = acc[mt][j][r];
        }
      }
  }
}

// --------- persistent sequential scan for one layer (single workgroup) ---------
// state_{t+1} = LN( 2*s + (tanh(U_t + s@Wh + b) - s)/tau ), written to hseq[t].
// U_t is staged into LDS one step ahead (TDM / async pipeline).
__global__ __launch_bounds__(256) void lnn_scan(
    const float* __restrict__ U,      // [T*B][H]
    float* __restrict__ hseq,         // [T*B][H] (overwritten with layer output)
    const float* __restrict__ Wh,     // [H][H]
    const float* __restrict__ bias, const float* __restrict__ tau_raw,
    const float* __restrict__ ln_g, const float* __restrict__ ln_b) {
  extern __shared__ char smem[];
  f16*   WhT = (f16*)smem;                              // [N][K] f16, 128 KB
  float* stF = (float*)(smem + 131072);                 // [64][256] f32 carry
  f16*   stH = (f16*)(smem + 131072 + 65536);           // [64][256] f16 copy
  float* prm = (float*)(smem + 131072 + 65536 + 32768); // bias | 1/tau | g | be
  float* Ub  = (float*)(smem + 233472);                 // [64][256] f32 U_t stage
  const int tid = threadIdx.x;

  for (int i = tid; i < Hsz*Hsz; i += 256) {
    int k = i >> 8, n = i & 255;
    WhT[n*Hsz + k] = (f16)Wh[i];
  }
  for (int i = tid; i < Bsz*Hsz; i += 256) { stF[i] = 0.0f; stH[i] = (f16)0.0f; }
  {
    int i = tid;                                        // H == blockDim.x == 256
    prm[i] = bias[i];
    float tr  = tau_raw[i];
    float tau = (tr > 20.0f) ? tr : log1pf(expf(tr));   // softplus
    prm[Hsz + i]   = 1.0f / tau;
    prm[2*Hsz + i] = ln_g[i];
    prm[3*Hsz + i] = ln_b[i];
  }
  // pre-stage U_0
  stage_U_issue(Ub, U, tid);
  stage_U_wait();
  __syncthreads();

  const int wv = tid >> 5, lane = tid & 31;
  const int mloc = lane & 15, kh = lane >> 4;
  const int n0 = wv * 32;

  for (int t = 0; t < Tsz; ++t) {
    // ---- phase A0: acc = U_t (LDS) + bias ----
    v8f acc[4][2];
#pragma unroll
    for (int mt = 0; mt < 4; ++mt)
#pragma unroll
      for (int j = 0; j < 2; ++j) {
        int n = n0 + j*16 + mloc;
        float bn = prm[n];
#pragma unroll
        for (int r = 0; r < 8; ++r) {
          int m = mt*16 + r + 8*kh;
          acc[mt][j][r] = Ub[m*Hsz + n] + bn;
        }
      }
    __syncthreads();   // everyone finished reading Ub

    // ---- kick off DMA of U_{t+1} into Ub (overlaps GEMM + LN) ----
    if (t + 1 < Tsz)
      stage_U_issue(Ub, U + (size_t)(t + 1)*Bsz*Hsz, tid);

    // ---- phase A1: acc += state @ Wh (WMMA f16, f32 accumulate) ----
#pragma unroll
    for (int kb = 0; kb < 8; ++kb) {
      const int kbase = kb * 32;
      v16h a[4], bf[2];
#pragma unroll
      for (int mt = 0; mt < 4; ++mt)
        a[mt] = loadA_frag(stH + (mt*16 + mloc)*Hsz, kbase + kh*8);
#pragma unroll
      for (int j = 0; j < 2; ++j)
        bf[j] = loadB_frag(WhT + (n0 + j*16 + mloc)*Hsz, kbase + kh*8);
#pragma unroll
      for (int mt = 0; mt < 4; ++mt)
#pragma unroll
        for (int j = 0; j < 2; ++j)
          acc[mt][j] = __builtin_amdgcn_wmma_f32_16x16x32_f16(
              false, a[mt], false, bf[j], (short)0, acc[mt][j], false, false);
    }

    // ---- phase B: liquid ODE update, stash residual into stF ----
#pragma unroll
    for (int mt = 0; mt < 4; ++mt)
#pragma unroll
      for (int j = 0; j < 2; ++j) {
        int n = n0 + j*16 + mloc;
        float itau = prm[Hsz + n];
#pragma unroll
        for (int r = 0; r < 8; ++r) {
          int m = mt*16 + r + 8*kh;
          float s  = stF[m*Hsz + n];
          float ns = s + (tanhf(acc[mt][j][r]) - s) * itau;
          stF[m*Hsz + n] = s + ns;          // residual (state + new_state)
        }
      }
    __syncthreads();   // GEMM stH reads + stF writes complete block-wide

    // ---- phase C: row LayerNorm over H, write new carry + layer output ----
    float* outp = hseq + (size_t)t * Bsz * Hsz;
    for (int r = 0; r < 8; ++r) {
      int row = wv*8 + r;
      float v[8], s = 0.0f, s2 = 0.0f;
#pragma unroll
      for (int i = 0; i < 8; ++i) {
        float xv = stF[row*Hsz + lane + 32*i];
        v[i] = xv; s += xv; s2 += xv*xv;
      }
#pragma unroll
      for (int o = 16; o >= 1; o >>= 1) {
        s  += __shfl_xor(s,  o, 32);
        s2 += __shfl_xor(s2, o, 32);
      }
      float mean = s * (1.0f/Hsz);
      float var  = s2 * (1.0f/Hsz) - mean*mean;
      float rstd = rsqrtf(var + 1e-5f);
#pragma unroll
      for (int i = 0; i < 8; ++i) {
        int c = lane + 32*i;
        float y = (v[i] - mean) * rstd * prm[2*Hsz + c] + prm[3*Hsz + c];
        stF[row*Hsz + c]  = y;
        stH[row*Hsz + c]  = (f16)y;
        outp[row*Hsz + c] = y;
      }
    }
    stage_U_wait();    // U_{t+1} landed in LDS (issuing wave)
    __syncthreads();   // all waves may now read Ub
  }
}

// ------------- domain gating + [T,B,H] -> [B,T,H] transpose -------------
__global__ __launch_bounds__(256) void lnn_gate(
    const float* __restrict__ hseq, const float* __restrict__ gate_w,
    const int* __restrict__ dom, float* __restrict__ hid) {
  size_t idx = (size_t)blockIdx.x*256 + threadIdx.x;   // ((b*T)+t)*H + h
  int h = (int)(idx & 255);
  size_t bt = idx >> 8;
  int b = (int)(bt >> 11), t = (int)(bt & 2047);
  float gw = gate_w[dom[0]*Hsz + h];
  float sg = 1.0f / (1.0f + expf(-gw));
  hid[idx] = hseq[((size_t)t*Bsz + b)*Hsz + h] * sg;
}

// ------------- predictions / anomaly / surprise (one wave per (b,t)) -------------
__global__ __launch_bounds__(256) void lnn_heads(
    const float* __restrict__ hid, const float* __restrict__ x,
    const float* __restrict__ ns_g, const float* __restrict__ ns_be,
    const float* __restrict__ ns_w, const float* __restrict__ ns_b,
    const float* __restrict__ an_g, const float* __restrict__ an_be,
    const float* __restrict__ an_w, const float* __restrict__ an_b,
    float* __restrict__ pred, float* __restrict__ anom, float* __restrict__ surp) {
  int row  = blockIdx.x*8 + (threadIdx.x >> 5);  // b*T + t
  int lane = threadIdx.x & 31;
  int b = row >> 11, t = row & 2047;
  const float* h = hid + (size_t)row * Hsz;

  float v[8], s = 0.0f, s2 = 0.0f;
#pragma unroll
  for (int i = 0; i < 8; ++i) {
    float xv = h[lane + 32*i];
    v[i] = xv; s += xv; s2 += xv*xv;
  }
#pragma unroll
  for (int o = 16; o >= 1; o >>= 1) { s += __shfl_xor(s, o, 32); s2 += __shfl_xor(s2, o, 32); }
  float mean = s * (1.0f/Hsz);
  float rstd = rsqrtf(s2 * (1.0f/Hsz) - mean*mean + 1e-5f);

  float accA = 0.0f, accP[8];
#pragma unroll
  for (int j = 0; j < 8; ++j) accP[j] = 0.0f;
#pragma unroll
  for (int i = 0; i < 8; ++i) {
    int c = lane + 32*i;
    float z  = (v[i] - mean) * rstd;
    accA    += (z*an_g[c] + an_be[c]) * an_w[c];
    float zn = z*ns_g[c] + ns_be[c];
#pragma unroll
    for (int j = 0; j < 8; ++j) accP[j] += zn * ns_w[c*INsz + j];
  }
#pragma unroll
  for (int o = 16; o >= 1; o >>= 1) accA += __shfl_xor(accA, o, 32);
  if (lane == 0)
    anom[(size_t)b*Tsz + t] = 1.0f / (1.0f + expf(-(accA + an_b[0])));

  if (t < Tsz - 1) {
#pragma unroll
    for (int j = 0; j < 8; ++j)
#pragma unroll
      for (int o = 16; o >= 1; o >>= 1) accP[j] += __shfl_xor(accP[j], o, 32);
    if (lane == 0) {
      size_t pbase = ((size_t)b*(Tsz-1) + t)*INsz;
      const float* xn = x + ((size_t)b*Tsz + (t+1))*INsz;
      float ss = 0.0f;
#pragma unroll
      for (int j = 0; j < 8; ++j) {
        float pj = accP[j] + ns_b[j];
        pred[pbase + j] = pj;
        float d = pj - xn[j];
        ss += d*d;
      }
      surp[(size_t)b*(Tsz-1) + t] = ss * (1.0f/INsz);
    }
  }
}

// ------------- properties head: LN -> 256x128 GELU MLP -> 128x2 -------------
__global__ __launch_bounds__(128) void lnn_props(
    const float* __restrict__ hid,
    const float* __restrict__ g, const float* __restrict__ be,
    const float* __restrict__ w1, const float* __restrict__ b1,
    const float* __restrict__ w2, const float* __restrict__ b2,
    float* __restrict__ props) {
  __shared__ float ph[Hsz];
  __shared__ float h1[Hsz/2];
  __shared__ float red[8];
  int b = blockIdx.x, tid = threadIdx.x;
  const float* h = hid + ((size_t)b*Tsz + (Tsz-1)) * Hsz;
  float x0 = h[tid], x1 = h[tid + 128];
  float s = x0 + x1, s2 = x0*x0 + x1*x1;
  int wv = tid >> 5, lane = tid & 31;
#pragma unroll
  for (int o = 16; o >= 1; o >>= 1) { s += __shfl_xor(s, o, 32); s2 += __shfl_xor(s2, o, 32); }
  if (lane == 0) { red[wv] = s; red[4 + wv] = s2; }
  __syncthreads();
  float S  = red[0] + red[1] + red[2] + red[3];
  float S2 = red[4] + red[5] + red[6] + red[7];
  float mean = S * (1.0f/Hsz);
  float rstd = rsqrtf(S2 * (1.0f/Hsz) - mean*mean + 1e-5f);
  ph[tid]       = (x0 - mean)*rstd*g[tid]       + be[tid];
  ph[tid + 128] = (x1 - mean)*rstd*g[tid + 128] + be[tid + 128];
  __syncthreads();
  float acc = b1[tid];
  for (int hh = 0; hh < Hsz; ++hh) acc += ph[hh] * w1[hh*(Hsz/2) + tid];
  h1[tid] = 0.5f * acc * (1.0f + erff(acc * 0.70710678118f));   // exact GELU
  __syncthreads();
  if (tid < NPsz) {
    float a = b2[tid];
    for (int j = 0; j < Hsz/2; ++j) a += h1[j] * w2[j*NPsz + tid];
    props[b*NPsz + tid] = a;
  }
}

// ------------------------------- launch -------------------------------
extern "C" void kernel_launch(void* const* d_in, const int* in_sizes, int n_in,
                              void* d_out, int out_size, void* d_ws, size_t ws_size,
                              hipStream_t stream) {
  const float* x       = (const float*)d_in[0];
  const float* in_w    = (const float*)d_in[1];
  const float* in_b    = (const float*)d_in[2];
  const float* Wx      = (const float*)d_in[3];
  const float* Wh      = (const float*)d_in[4];
  const float* bvec    = (const float*)d_in[5];
  const float* tau_raw = (const float*)d_in[6];
  const float* ln_g    = (const float*)d_in[7];
  const float* ln_b    = (const float*)d_in[8];
  const float* gate_w  = (const float*)d_in[9];
  const float* ns_ln_g = (const float*)d_in[10];
  const float* ns_ln_b = (const float*)d_in[11];
  const float* ns_w    = (const float*)d_in[12];
  const float* ns_b    = (const float*)d_in[13];
  const float* an_ln_g = (const float*)d_in[14];
  const float* an_ln_b = (const float*)d_in[15];
  const float* an_w    = (const float*)d_in[16];
  const float* an_b    = (const float*)d_in[17];
  const float* pr_ln_g = (const float*)d_in[18];
  const float* pr_ln_b = (const float*)d_in[19];
  const float* pr_w1   = (const float*)d_in[20];
  const float* pr_b1   = (const float*)d_in[21];
  const float* pr_w2   = (const float*)d_in[22];
  const float* pr_b2   = (const float*)d_in[23];
  const int*   dom     = (const int*)d_in[24];

  float* out  = (float*)d_out;
  float* hseq = (float*)d_ws;              // [T,B,H] f32, 128 MB
  float* U    = out + HID_OFF;             // reuse hidden region as U scratch

  const size_t need_ws = (size_t)Tsz*Bsz*Hsz*sizeof(float);
  if (ws_size < need_ws) return;           // cannot run without scratch

  const int gemm_smem = 131072 + 32768;                 // WxT + Ah
  const int scan_smem = 233472 + Bsz*Hsz*(int)sizeof(float);  // + Ub = 299008 B

  lnn_inproj<<<Tsz*Bsz, 256, 0, stream>>>(x, in_w, in_b, hseq);

  for (int l = 0; l < Lsz; ++l) {
    lnn_gemm_hWx<<<(Tsz*Bsz)/(64*GCHUNK), 256, gemm_smem, stream>>>(
        hseq, Wx + (size_t)l*Hsz*Hsz, U);
    lnn_scan<<<1, 256, scan_smem, stream>>>(
        U, hseq, Wh + (size_t)l*Hsz*Hsz,
        bvec + l*Hsz, tau_raw + l*Hsz, ln_g + l*Hsz, ln_b + l*Hsz);
  }

  lnn_gate<<<(int)(((size_t)Bsz*Tsz*Hsz)/256), 256, 0, stream>>>(
      hseq, gate_w, dom, out + HID_OFF);

  lnn_heads<<<(Bsz*Tsz)/8, 256, 0, stream>>>(
      out + HID_OFF, x,
      ns_ln_g, ns_ln_b, ns_w, ns_b,
      an_ln_g, an_ln_b, an_w, an_b,
      out + PRED_OFF, out + ANOM_OFF, out + SURP_OFF);

  lnn_props<<<Bsz, 128, 0, stream>>>(
      out + HID_OFF, pr_ln_g, pr_ln_b, pr_w1, pr_b1, pr_w2, pr_b2,
      out + PROP_OFF);
}
